// GATLayer_46256797778528
// MI455X (gfx1250) — matrix operand order repaired and checked by reading, not compiled, not run on hardware
//
#include <hip/hip_runtime.h>
#include <hip/hip_bf16.h>

typedef __attribute__((ext_vector_type(16))) _Float16 v16h;
typedef __attribute__((ext_vector_type(8)))  float    v8f;

#define IN_F   128
#define HC     128      // H*C
#define NHEAD  4
#define CCH    32
#define ALPHA_LRELU 0.2f

// ---------- helpers: order-preserving float <-> uint key for atomic max ----------
__device__ __forceinline__ unsigned fkey(float f) {
    unsigned u = __float_as_uint(f);
    return (u & 0x80000000u) ? ~u : (u | 0x80000000u);
}
__device__ __forceinline__ float kdec(unsigned k) {
    unsigned u = (k & 0x80000000u) ? (k ^ 0x80000000u) : ~k;
    return __uint_as_float(u);
}

// ---------- zero-fill 32-bit words ----------
__global__ void zero_u32(unsigned* __restrict__ p, long long n) {
    long long i = (long long)blockIdx.x * blockDim.x + threadIdx.x;
    if (i < n) p[i] = 0u;
}

// ---------- f32 -> f16 conversion ----------
__global__ void cvt_f16(const float* __restrict__ src, _Float16* __restrict__ dst, int n) {
    int i = blockIdx.x * blockDim.x + threadIdx.x;
    if (i < n) dst[i] = (_Float16)src[i];
}

// ---------- h = x @ W^T + b   via v_wmma_f32_16x16x32_f16 ----------
// One wave per 16x16 output tile. Block = 256 threads = 8 waves = 8 N-tiles (128 cols).
// grid.x = N/16 (N divisible by 16 for this problem: 50000 = 3125*16).
__global__ void gemm_wmma(const _Float16* __restrict__ X,    // [N,128] f16
                          const _Float16* __restrict__ Wm,   // [128,128] f16 (row j = output col j)
                          const float*    __restrict__ bias, // [128]
                          float*          __restrict__ Hh,   // [N,128] f32 out
                          int Nn) {
    const int wave = threadIdx.x >> 5;
    const int lane = threadIdx.x & 31;
    const int m0 = blockIdx.x * 16;
    const int n0 = wave * 16;
    if (m0 >= Nn) return;

    const int  r  = lane & 15;
    const bool hi = lane >= 16;

    // C/D layout: per lane, N is fixed (= n0 + r); M = elem + (hi ? 8 : 0).
    v8f acc;
    const float bv = bias[n0 + r];
#pragma unroll
    for (int i = 0; i < 8; ++i) acc[i] = bv;

    // A: row (m0+r); lanes 0-15 cover K {0..7,16..23}, lanes 16-31 cover K {8..15,24..31}
    const _Float16* xrow = X  + (size_t)(m0 + r) * IN_F + (hi ? 8 : 0);
    // B[k][n] = W[n0+n][k]; lanes 0-15 K {0..7,16..23}, lanes 16-31 K {8..15,24..31}, column = lane&15
    const _Float16* wrow = Wm + (size_t)(n0 + r) * IN_F + (hi ? 8 : 0);

    for (int kk = 0; kk < IN_F; kk += 32) {
        v16h afrag, bfrag;
#pragma unroll
        for (int i = 0; i < 8; ++i) {
            afrag[i]     = xrow[kk + i];
            afrag[i + 8] = xrow[kk + 16 + i];
        }
#pragma unroll
        for (int i = 0; i < 16; ++i) bfrag[i] = wrow[kk + (i < 8 ? i : i + 8)];

        acc = __builtin_amdgcn_wmma_f32_16x16x32_f16(
            /*neg_a=*/false, afrag, /*neg_b=*/false, bfrag,
            /*c_mod=*/(short)0, acc, /*reuse_a=*/false, /*reuse_b=*/false);
    }

    const int col   = n0 + r;
    const int mbase = m0 + (hi ? 8 : 0);
#pragma unroll
    for (int i = 0; i < 8; ++i)
        Hh[(size_t)(mbase + i) * HC + col] = acc[i];
}

// ---------- per-node attention coefficients ----------
// alpha_src[n,h] = <h[n,h,:], a[h,0:32]> ; alpha_dst[n,h] = <h[n,h,:], a[h,32:64]>
__global__ void node_alpha(const float* __restrict__ Hh, const float* __restrict__ A,
                           float* __restrict__ as_, float* __restrict__ ad_, int Nn) {
    int idx = blockIdx.x * blockDim.x + threadIdx.x;   // n*4 + h
    if (idx >= Nn * NHEAD) return;
    int n = idx >> 2, h = idx & 3;
    const float* hp = Hh + (size_t)n * HC + h * CCH;
    const float* ap = A + h * (2 * CCH);
    float s = 0.f, d = 0.f;
#pragma unroll
    for (int c = 0; c < CCH; ++c) {
        float v = hp[c];
        s += v * ap[c];
        d += v * ap[CCH + c];
    }
    as_[idx] = s;
    ad_[idx] = d;
}

__device__ __forceinline__ float edge_logit(const float* as_, const float* ad_,
                                            int s, int d, int h) {
    float l = as_[s * NHEAD + h] + ad_[d * NHEAD + h];
    return l > 0.f ? l : ALPHA_LRELU * l;
}

// ---------- pass 1: segment max over src (atomic max on monotone uint key) ----------
__global__ void edge_max(const int* __restrict__ edges, const float* __restrict__ as_,
                         const float* __restrict__ ad_, unsigned* __restrict__ mkey, int E) {
    int idx = blockIdx.x * blockDim.x + threadIdx.x;   // e*4 + h
    if (idx >= E * NHEAD) return;
    int e = idx >> 2, h = idx & 3;
    int s = edges[2 * e], d = edges[2 * e + 1];
    float l = edge_logit(as_, ad_, s, d, h);
    atomicMax(&mkey[s * NHEAD + h], fkey(l));
}

// ---------- pass 2: denom[src,h] += exp(l - m[src,h]) ----------
__global__ void edge_denom(const int* __restrict__ edges, const float* __restrict__ as_,
                           const float* __restrict__ ad_, const unsigned* __restrict__ mkey,
                           float* __restrict__ denom, int E) {
    int idx = blockIdx.x * blockDim.x + threadIdx.x;
    if (idx >= E * NHEAD) return;
    int e = idx >> 2, h = idx & 3;
    int s = edges[2 * e], d = edges[2 * e + 1];
    float l  = edge_logit(as_, ad_, s, d, h);
    float m  = kdec(mkey[s * NHEAD + h]);
    atomicAdd(&denom[s * NHEAD + h], __expf(l - m));
}

// ---------- pass 3: out[src] += softmax_prob * h[dst]  (thread per edge-channel) ----------
__global__ void edge_agg(const int* __restrict__ edges, const float* __restrict__ as_,
                         const float* __restrict__ ad_, const unsigned* __restrict__ mkey,
                         const float* __restrict__ denom, const float* __restrict__ Hh,
                         float* __restrict__ out, long long total) {
    long long idx = (long long)blockIdx.x * blockDim.x + threadIdx.x; // e*128 + t
    if (idx >= total) return;
    int t = (int)(idx & (HC - 1));
    int e = (int)(idx >> 7);
    int h = t >> 5;
    int s = edges[2 * e], d = edges[2 * e + 1];
    float l = edge_logit(as_, ad_, s, d, h);
    float m = kdec(mkey[s * NHEAD + h]);
    float p = __expf(l - m) / denom[s * NHEAD + h];
    atomicAdd(&out[(size_t)s * HC + t], p * Hh[(size_t)d * HC + t]);
}

extern "C" void kernel_launch(void* const* d_in, const int* in_sizes, int n_in,
                              void* d_out, int out_size, void* d_ws, size_t ws_size,
                              hipStream_t stream) {
    const float* x     = (const float*)d_in[0];   // [N,128]
    const float* W     = (const float*)d_in[1];   // [128,128]
    const float* b     = (const float*)d_in[2];   // [128]
    const float* a     = (const float*)d_in[3];   // [4,64]
    const int*   edges = (const int*)d_in[4];     // [E,2]

    const int Nn = in_sizes[0] / IN_F;   // 50000
    const int E  = in_sizes[4] / 2;      // 500000
    float* out = (float*)d_out;

    // ---- carve workspace ----
    char* ws = (char*)d_ws;
    size_t off = 0;
    auto carve = [&](size_t bytes) -> void* {
        void* p = ws + off;
        off = (off + bytes + 255) & ~(size_t)255;
        return p;
    };
    float*     Hh    = (float*)    carve((size_t)Nn * HC * sizeof(float));
    _Float16*  X16   = (_Float16*) carve((size_t)Nn * IN_F * sizeof(_Float16));
    _Float16*  W16   = (_Float16*) carve((size_t)HC * IN_F * sizeof(_Float16));
    float*     as_   = (float*)    carve((size_t)Nn * NHEAD * sizeof(float));
    float*     ad_   = (float*)    carve((size_t)Nn * NHEAD * sizeof(float));
    unsigned*  mkey  = (unsigned*) carve((size_t)Nn * NHEAD * sizeof(unsigned));
    float*     denom = (float*)    carve((size_t)Nn * NHEAD * sizeof(float));

    const int B = 256;

    // 0) init accumulators + output (harness poisons them)
    {
        long long n = (long long)Nn * HC;
        zero_u32<<<(int)((n + B - 1) / B), B, 0, stream>>>((unsigned*)out, n);
        long long m = (long long)Nn * NHEAD;
        zero_u32<<<(int)((m + B - 1) / B), B, 0, stream>>>(mkey, m);
        zero_u32<<<(int)((m + B - 1) / B), B, 0, stream>>>((unsigned*)denom, m);
    }

    // 1) f32 -> f16 for WMMA operands
    {
        int nx = Nn * IN_F;
        cvt_f16<<<(nx + B - 1) / B, B, 0, stream>>>(x, X16, nx);
        int nw = HC * IN_F;
        cvt_f16<<<(nw + B - 1) / B, B, 0, stream>>>(W, W16, nw);
    }

    // 2) h = x @ W^T + b  (WMMA; N divisible by 16)
    gemm_wmma<<<Nn / 16, B, 0, stream>>>(X16, W16, b, Hh, Nn);

    // 3) per-node attention coefficients
    node_alpha<<<(Nn * NHEAD + B - 1) / B, B, 0, stream>>>(Hh, a, as_, ad_, Nn);

    // 4) segment max
    edge_max<<<(E * NHEAD + B - 1) / B, B, 0, stream>>>(edges, as_, ad_, mkey, E);

    // 5) segment sum of exp
    edge_denom<<<(E * NHEAD + B - 1) / B, B, 0, stream>>>(edges, as_, ad_, mkey, denom, E);

    // 6) weighted aggregation
    {
        long long total = (long long)E * HC;
        edge_agg<<<(int)((total + B - 1) / B), B, 0, stream>>>(
            edges, as_, ad_, mkey, denom, Hh, out, total);
    }
}
